// EncoderVIG_V20_2_72550587564465
// MI455X (gfx1250) — compile-verified, matrix operand order checked by reading, not run on hardware
//
#include <hip/hip_runtime.h>
#include <hip/hip_bf16.h>
#include <math.h>

// ---------------------------------------------------------------------------
// CDNA5 (gfx1250) ViG encoder.
//  * All 1x1 convs + pairwise-distance einsum -> bf16 WMMA GEMM
//    (v_wmma_f32_16x16x32_bf16, f32 accumulate), fused BN/GELU/res/dist.
//  * Double-buffered LDS pipeline: tile kt+1 staged from global into registers
//    before the WMMA on tile kt; single barrier per K-tile.
//  * 256 threads = 8 waves (wave32), 2x4 wave grid, 32x64 macro tile.
//  * LDS tiles fragment-major: each lane's 16-bf16 WMMA fragment is 32
//    contiguous bytes -> ds_load_b128 pairs, no scalar ds_load_u16.
//  * Activations carried as bf16 (producers dual-store f32+bf16).
//  * gather_agg stages neighbor indices via GLOBAL_LOAD_ASYNC_TO_LDS +
//    s_wait_asynccnt (builtins confirmed present by earlier rounds).
// ---------------------------------------------------------------------------

typedef __attribute__((ext_vector_type(16))) __bf16 v16bf;
typedef __attribute__((ext_vector_type(8)))  __bf16 v8bf;
typedef __attribute__((ext_vector_type(4)))  __bf16 v4bf;
typedef __attribute__((ext_vector_type(8)))  float  v8f;
typedef __attribute__((ext_vector_type(4)))  float  vf4;

#if defined(__HIP_DEVICE_COMPILE__) && __has_builtin(__builtin_amdgcn_global_load_async_to_lds_b32)
#define HAVE_ASYNC_LDS 1
#else
#define HAVE_ASYNC_LDS 0
#endif

#if defined(__HIP_DEVICE_COMPILE__) && __has_builtin(__builtin_amdgcn_s_wait_asynccnt)
#define WAIT_ASYNC() __builtin_amdgcn_s_wait_asynccnt(0)
#else
#define WAIT_ASYNC() asm volatile("s_wait_asynccnt 0x0" ::: "memory")
#endif

__device__ __forceinline__ float gelu_tanh(float x) {
  const float k0 = 0.7978845608028654f;   // sqrt(2/pi)
  const float k1 = 0.044715f;
  float x3 = x * x * x;
  return 0.5f * x * (1.0f + tanhf(k0 * (x + k1 * x3)));
}

// GEMM epilogue mode bits
#define GM_BN   1
#define GM_GELU 2
#define GM_RES  4
#define GM_DIST 8

// Fragment-major LDS slot: for K index k within a 32-deep tile,
//   half = (k>>3)&1 ; elem = ((k>>4)<<3)|(k&7)
// slot(row, half) holds 16 contiguous bf16 = one lane's WMMA fragment.
#define FRAG_BLK 40                    // bf16 units per slot, 80B stride
#define A_STRIDE (64 * FRAG_BLK)       // one A buffer: 32 rows x 2 halves
#define B_STRIDE (128 * FRAG_BLK)      // one B buffer: 64 cols x 2 halves

// D[M,N] = A[M,K](f32) * B[K,N](bf16); f32 accumulate in WMMA.
// 256 threads = 8 waves, 2x4 wave grid, 32x64 macro tile, K step 32,
// double-buffered LDS.
template <int MODE>
__global__ __launch_bounds__(256) void gemm_wmma_bf16(
    const float* __restrict__ A, long a_bs, long lda_m, long lda_k,
    const __bf16* __restrict__ Bm, long b_bs, long ldb_k, long ldb_n,
    int Mrows, int Ncols, int Ktot,
    float* __restrict__ O, long o_bs, long ldo,
    __bf16* __restrict__ Obf,
    const float* __restrict__ bias, const float* __restrict__ gsc,
    const float* __restrict__ bsh, const float* __restrict__ res,
    const float* __restrict__ rel)
{
  __shared__ __bf16 As[2 * A_STRIDE];
  __shared__ __bf16 Bs[2 * B_STRIDE];

  const int tid  = threadIdx.x;
  const int lane = tid & 31;
  const int wave = tid >> 5;
  const int wm   = wave >> 2;       // 0..1
  const int wn   = wave & 3;        // 0..3
  const int hsel = lane >> 4;       // lane half per ISA 16-bit layout
  const int l16  = lane & 15;
  const int bz   = blockIdx.z;
  const int m0   = blockIdx.y * 32;
  const int n0   = blockIdx.x * 64;

  const float*  Ab = A  + (size_t)bz * a_bs;
  const __bf16* Bb = Bm + (size_t)bz * b_bs;

  const int ktiles = (Ktot + 31) >> 5;

  float  aR[4];
  __bf16 bR[8];

  // ---- stage one K-tile (A: 32x32 f32, B: 32x64 bf16) into registers ----
  auto loadTile = [&](int kt) {
    const int k0 = kt << 5;
    const bool fullK = (k0 + 32 <= Ktot);
    // A: 4 elems/thread
    if (fullK && (m0 + 32 <= Mrows) && lda_k == 1) {
      int r = tid >> 3, seg = (tid & 7) * 4;
      vf4 f = *(const vf4*)(Ab + (size_t)(m0 + r) * lda_m + (k0 + seg));
#pragma unroll
      for (int j = 0; j < 4; ++j) aR[j] = f[j];
    } else if (fullK && (m0 + 32 <= Mrows) && lda_m == 1) {
      int kk = tid >> 3, seg = (tid & 7) * 4;
      vf4 f = *(const vf4*)(Ab + (size_t)(m0 + seg) + (size_t)(k0 + kk) * lda_k);
#pragma unroll
      for (int j = 0; j < 4; ++j) aR[j] = f[j];
    } else {
#pragma unroll
      for (int i = 0; i < 4; ++i) {
        int e = tid + i * 256;
        int r = e >> 5, c = e & 31;
        float av = 0.f;
        if ((m0 + r) < Mrows && (k0 + c) < Ktot)
          av = Ab[(size_t)(m0 + r) * lda_m + (size_t)(k0 + c) * lda_k];
        aR[i] = av;
      }
    }
    // B: 8 elems/thread (ldb_n==1 in all uses; Ncols multiple of 64)
    if (fullK && (n0 + 64 <= Ncols) && ldb_n == 1) {
      int kk = tid >> 3, seg = (tid & 7) * 8;
      v8bf hb = *(const v8bf*)(Bb + (size_t)(k0 + kk) * ldb_k + (n0 + seg));
#pragma unroll
      for (int j = 0; j < 8; ++j) bR[j] = hb[j];
    } else {
#pragma unroll
      for (int i = 0; i < 8; ++i) {
        int e = tid + i * 256;
        int r = e >> 6, c = e & 63;        // r = k, c = n
        __bf16 bv = (__bf16)0.f;
        if ((k0 + r) < Ktot && (n0 + c) < Ncols)
          bv = Bb[(size_t)(k0 + r) * ldb_k + (size_t)(n0 + c) * ldb_n];
        bR[i] = bv;
      }
    }
  };

  // ---- commit staged registers into fragment-major LDS buffer `buf` ----
  auto storeTile = [&](int kt, int buf) {
    const int k0 = kt << 5;
    const bool fullK = (k0 + 32 <= Ktot);
    __bf16* Ad = As + buf * A_STRIDE;
    __bf16* Bd = Bs + buf * B_STRIDE;
    if (fullK && (m0 + 32 <= Mrows) && lda_k == 1) {
      int r = tid >> 3, seg = (tid & 7) * 4;
      int hh = (seg >> 3) & 1;
      int eb = ((seg >> 4) << 3) | (seg & 7);
      v4bf hb;
#pragma unroll
      for (int j = 0; j < 4; ++j) hb[j] = (__bf16)aR[j];
      *(v4bf*)&Ad[(r * 2 + hh) * FRAG_BLK + eb] = hb;
    } else if (fullK && (m0 + 32 <= Mrows) && lda_m == 1) {
      int kk = tid >> 3, seg = (tid & 7) * 4;
      int hh = (kk >> 3) & 1;
      int el = ((kk >> 4) << 3) | (kk & 7);
#pragma unroll
      for (int j = 0; j < 4; ++j)
        Ad[((seg + j) * 2 + hh) * FRAG_BLK + el] = (__bf16)aR[j];
    } else {
#pragma unroll
      for (int i = 0; i < 4; ++i) {
        int e = tid + i * 256;
        int r = e >> 5, c = e & 31;
        int hh = (c >> 3) & 1;
        int el = ((c >> 4) << 3) | (c & 7);
        Ad[(r * 2 + hh) * FRAG_BLK + el] = (__bf16)aR[i];
      }
    }
    if (fullK && (n0 + 64 <= Ncols) && ldb_n == 1) {
      int kk = tid >> 3, seg = (tid & 7) * 8;
      int hh = (kk >> 3) & 1;
      int el = ((kk >> 4) << 3) | (kk & 7);
#pragma unroll
      for (int j = 0; j < 8; ++j)
        Bd[((seg + j) * 2 + hh) * FRAG_BLK + el] = bR[j];
    } else {
#pragma unroll
      for (int i = 0; i < 8; ++i) {
        int e = tid + i * 256;
        int r = e >> 6, c = e & 63;
        int hh = (r >> 3) & 1;
        int el = ((r >> 4) << 3) | (r & 7);
        Bd[(c * 2 + hh) * FRAG_BLK + el] = bR[i];
      }
    }
  };

  v8f acc = {};

  // prologue: stage tile 0
  loadTile(0);
  storeTile(0, 0);
  __syncthreads();

  for (int kt = 0; kt < ktiles; ++kt) {
    const int cur = kt & 1;
    const bool more = (kt + 1 < ktiles);
    if (more) {
      loadTile(kt + 1);                 // global loads overlap the WMMA below
      if (kt + 2 < ktiles)
        __builtin_prefetch(Bb + (size_t)((kt + 2) << 5) * ldb_k + n0 + (tid >> 2),
                           0, 1);
    }

    // per-lane fragments: contiguous 32B vector ds loads
    const int ablk = cur * A_STRIDE + ((wm * 16 + l16) * 2 + hsel) * FRAG_BLK;
    const int bblk = cur * B_STRIDE + ((wn * 16 + l16) * 2 + hsel) * FRAG_BLK;
    v8bf alo = *(const v8bf*)&As[ablk];
    v8bf ahi = *(const v8bf*)&As[ablk + 8];
    v8bf blo = *(const v8bf*)&Bs[bblk];
    v8bf bhi = *(const v8bf*)&Bs[bblk + 8];
    v16bf af = __builtin_shufflevector(alo, ahi, 0, 1, 2, 3, 4, 5, 6, 7,
                                       8, 9, 10, 11, 12, 13, 14, 15);
    v16bf bf = __builtin_shufflevector(blo, bhi, 0, 1, 2, 3, 4, 5, 6, 7,
                                       8, 9, 10, 11, 12, 13, 14, 15);
    acc = __builtin_amdgcn_wmma_f32_16x16x32_bf16(false, af, false, bf,
                                                  (short)0, acc, false, false);
    if (more) {
      storeTile(kt + 1, cur ^ 1);
      __syncthreads();
    }
  }

  // Epilogue: C/D layout -> VGPR j : M = j + 8*half, N = lane&15
  float*  Ob  = O   + (size_t)bz * o_bs;
  __bf16* Obb = Obf ? (Obf + (size_t)bz * o_bs) : nullptr;
#pragma unroll
  for (int j = 0; j < 8; ++j) {
    int m = m0 + wm * 16 + hsel * 8 + j;
    int n = n0 + wn * 16 + l16;
    if (m < Mrows && n < Ncols) {
      float v = acc[j];
      if (MODE & GM_BN)   v = (v + bias[m]) * gsc[m] + bsh[m];
      if (MODE & GM_GELU) v = gelu_tanh(v);
      if (MODE & GM_DIST) v = 2.f * v - 2.f + rel[(size_t)m * ldo + n];
      size_t oi = (size_t)m * ldo + n;
      if (MODE & GM_RES)  v += res[(size_t)bz * o_bs + oi];
      Ob[oi] = v;
      if (Obb) Obb[oi] = (__bf16)v;
    }
  }
}

// Direct 3x3 conv (stem + downsamples), fused BN+GELU, optional bf16 twin.
__global__ void conv3x3_bn(const float* __restrict__ x, const float* __restrict__ w,
                           const float* __restrict__ bias, const float* __restrict__ g,
                           const float* __restrict__ be, float* __restrict__ out,
                           __bf16* __restrict__ outb,
                           int Bn, int Cin, int Hin, int Win, int Cout,
                           int Ho, int Wo, int stride, int act)
{
  size_t t = (size_t)blockIdx.x * blockDim.x + threadIdx.x;
  size_t tot = (size_t)Bn * Cout * Ho * Wo;
  if (t >= tot) return;
  int wo = (int)(t % Wo);
  int ho = (int)((t / Wo) % Ho);
  int co = (int)((t / ((size_t)Wo * Ho)) % Cout);
  int b  = (int)(t / ((size_t)Wo * Ho * Cout));
  float s = 0.f;
  for (int ci = 0; ci < Cin; ++ci) {
    const float* xp = x + ((size_t)(b * Cin + ci) * Hin) * Win;
    const float* wp = w + ((size_t)(co * Cin + ci)) * 9;
#pragma unroll
    for (int kh = 0; kh < 3; ++kh) {
      int hi = ho * stride + kh - 1;
      if (hi < 0 || hi >= Hin) continue;
#pragma unroll
      for (int kw = 0; kw < 3; ++kw) {
        int wi = wo * stride + kw - 1;
        if (wi < 0 || wi >= Win) continue;
        s += xp[(size_t)hi * Win + wi] * wp[kh * 3 + kw];
      }
    }
  }
  float v = (s + bias[co]) * g[co] + be[co];
  if (act) v = gelu_tanh(v);
  out[t] = v;
  if (outb) outb[t] = (__bf16)v;
}

__global__ void add_bcast(float* __restrict__ h, __bf16* __restrict__ hb,
                          const float* __restrict__ pos, size_t per, size_t tot)
{
  size_t t = (size_t)blockIdx.x * blockDim.x + threadIdx.x;
  if (t < tot) {
    float v = h[t] + pos[t % per];
    h[t] = v;
    hb[t] = (__bf16)v;
  }
}

__global__ void avgpool_rr(const float* __restrict__ x, float* __restrict__ o,
                           int Bn, int C, int H, int W, int r)
{
  int W2 = W / r, H2 = H / r;
  size_t tot = (size_t)Bn * C * H2 * W2;
  size_t t = (size_t)blockIdx.x * blockDim.x + threadIdx.x;
  if (t >= tot) return;
  int mx = (int)(t % W2);
  int my = (int)((t / W2) % H2);
  int c  = (int)((t / ((size_t)W2 * H2)) % C);
  int b  = (int)(t / ((size_t)W2 * H2 * C));
  const float* xp = x + ((size_t)(b * C + c) * H) * W;
  float s = 0.f;
  for (int dy = 0; dy < r; ++dy)
    for (int dx = 0; dx < r; ++dx)
      s += xp[(size_t)(my * r + dy) * W + (mx * r + dx)];
  o[t] = s / (float)(r * r);
}

// Column L2 normalization over channels; optional f32 and/or bf16 outputs.
__global__ void l2norm_cols(const float* __restrict__ x, float* __restrict__ of,
                            __bf16* __restrict__ ob, int Bn, int C, int N)
{
  size_t t = (size_t)blockIdx.x * blockDim.x + threadIdx.x;
  if (t >= (size_t)Bn * N) return;
  int n = (int)(t % N);
  int b = (int)(t / N);
  const float* xp = x + (size_t)b * C * N + n;
  float s = 0.f;
  for (int c = 0; c < C; ++c) { float v = xp[(size_t)c * N]; s += v * v; }
  float rs = rsqrtf(fmaxf(s, 1e-24f));
  for (int c = 0; c < C; ++c) {
    float v = xp[(size_t)c * N] * rs;
    size_t oi = (size_t)b * C * N + (size_t)c * N + n;
    if (of) of[oi] = v;
    if (ob) ob[oi] = (__bf16)v;
  }
}

// Per-row top-(9*dil) keeping every dil-th pick; wave32 per row, register-held
// row (M <= 256), argmax via __shfl_xor width 32.
__global__ __launch_bounds__(256) void topk_rows(const float* __restrict__ dist,
                                                 int rows, int Mp, int dil,
                                                 int* __restrict__ idxout)
{
  int lane = threadIdx.x & 31;
  int wv   = threadIdx.x >> 5;
  int row  = blockIdx.x * 8 + wv;
  if (row >= rows) return;
  const float* d = dist + (size_t)row * Mp;
  float v[8];
#pragma unroll
  for (int j = 0; j < 8; ++j) {
    int e = lane + j * 32;
    v[j] = (e < Mp) ? d[e] : -3.0e38f;
  }
  int ksel = 9 * dil;
  for (int s = 0; s < ksel; ++s) {
    float bv = -3.0e38f; int bi = 0x7fffffff;
#pragma unroll
    for (int j = 0; j < 8; ++j)
      if (v[j] > bv) { bv = v[j]; bi = j * 32 + lane; }
#pragma unroll
    for (int off = 16; off > 0; off >>= 1) {
      float ov = __shfl_xor(bv, off, 32);
      int   oi = __shfl_xor(bi, off, 32);
      if (ov > bv || (ov == bv && oi < bi)) { bv = ov; bi = oi; }
    }
    if ((s % dil) == 0 && lane == 0) idxout[(size_t)row * 9 + s / dil] = bi;
    if ((bi & 31) == lane) v[bi >> 5] = -3.0e38f;
  }
}

// Build gconv input (bf16): interleave [xf, max_j(yf[:,idx_j]-xf)] -> (B,2C,N).
// Neighbor index rows staged through LDS via GLOBAL_LOAD_ASYNC_TO_LDS.
__global__ __launch_bounds__(256) void gather_agg(
    const float* __restrict__ xf, const float* __restrict__ yf,
    const int* __restrict__ nn, __bf16* __restrict__ o,
    int C, int N, int Mp)
{
  __shared__ int sIdx[32 * 9];
  const int tid = threadIdx.x;
  const int b   = blockIdx.z;
  const int n0  = blockIdx.x * 32;
  const int* gip = nn + ((size_t)b * N + n0) * 9;
#if HAVE_ASYNC_LDS
  for (int u = tid; u < 288; u += 256)
    __builtin_amdgcn_global_load_async_to_lds_b32(
        (int*)(gip + u), sIdx + u, 0, 0);
  WAIT_ASYNC();
#else
  for (int u = tid; u < 288; u += 256) sIdx[u] = gip[u];
#endif
  __syncthreads();
  const int n  = n0 + (tid & 31);
  const int c0 = tid >> 5;                       // 8 channels per pass
  const int* ip = sIdx + (tid & 31) * 9;
  for (int c = c0; c < C; c += 8) {
    float xv = xf[((size_t)b * C + c) * N + n];
    const float* yp = yf + ((size_t)b * C + c) * Mp;
    float mx = -3.0e38f;
#pragma unroll
    for (int j = 0; j < 9; ++j) mx = fmaxf(mx, yp[ip[j]] - xv);
    __bf16* ob = o + (size_t)b * (size_t)(2 * C) * N;
    ob[(size_t)(2 * c) * N + n]     = (__bf16)xv;
    ob[(size_t)(2 * c + 1) * N + n] = (__bf16)mx;
  }
}

// ---------------------------------------------------------------------------
// Host orchestration
// ---------------------------------------------------------------------------

extern "C" void kernel_launch(void* const* d_in, const int* in_sizes, int n_in,
                              void* d_out, int out_size, void* d_ws, size_t ws_size,
                              hipStream_t stream)
{
  (void)in_sizes; (void)n_in; (void)out_size; (void)ws_size;
  int p = 0;
  auto F = [&](int i) { return (const float*)d_in[i]; };
  struct CP { const float *w, *b, *g, *be; };
  auto getCP = [&]() { CP c{F(p), F(p + 1), F(p + 2), F(p + 3)}; p += 4; return c; };

  const float* X = F(p++);
  CP st0 = getCP(), st1 = getCP(), st2 = getCP();
  const float* pos = F(p++);

  // Workspace layout
  const size_t M1 = (size_t)1024 * 1024;
  float* W0  = (float*)d_ws;
  float* A0  = W0;                 // 8M f32
  float* A1  = A0 + 8 * M1;        // 8M
  float* S1  = A1 + 8 * M1;        // 8M  xf / h'
  float* S2  = S1 + 8 * M1;        // 8M  ffn mid (f32 view)
  float* S3  = S2 + 8 * M1;        // 2M  pooled yf
  float* S4  = S3 + 2 * M1;        // 2M  xn (f32, GEMM A)
  float* Dst = S4 + 2 * M1;        // 9M  distance matrix
  int*   Idx = (int*)(Dst + 9 * M1);             // 0.5M ints
  __bf16* BFA = (__bf16*)(W0 + 46 * M1);         // bf16 arena
  __bf16* Hb0 = BFA;               // 2M bf16 twin of A0
  __bf16* Hb1 = Hb0 + 2 * M1;      // 2M twin of A1
  __bf16* S1b = Hb1 + 2 * M1;      // 2M twin of S1 (h')
  __bf16* S2b = S1b + 2 * M1;      // 8M twin of S2 (gather out / ffn mid)
  __bf16* Gb  = S2b + 8 * M1;      // 4M gconv bf16 out
  __bf16* S5b = Gb  + 4 * M1;      // 1M yn (bf16 GEMM B)

  const int Bn = 8;

  auto gemm = [&](int mode,
                  const float* A, long a_bs, long lda_m, long lda_k,
                  const __bf16* Bm, long b_bs, long ldb_k, long ldb_n,
                  int M, int Nc, int K, float* O, long o_bs, long ldo,
                  __bf16* Obf,
                  const float* bias, const float* g, const float* be,
                  const float* res, const float* rel) {
    dim3 gr((Nc + 63) / 64, (M + 31) / 32, Bn);
    dim3 bl(256);
    switch (mode) {
      case GM_BN:
        gemm_wmma_bf16<GM_BN><<<gr, bl, 0, stream>>>(A, a_bs, lda_m, lda_k, Bm, b_bs,
            ldb_k, ldb_n, M, Nc, K, O, o_bs, ldo, Obf, bias, g, be, res, rel);
        break;
      case GM_BN | GM_GELU:
        gemm_wmma_bf16<GM_BN | GM_GELU><<<gr, bl, 0, stream>>>(A, a_bs, lda_m, lda_k,
            Bm, b_bs, ldb_k, ldb_n, M, Nc, K, O, o_bs, ldo, Obf, bias, g, be, res, rel);
        break;
      case GM_BN | GM_RES:
        gemm_wmma_bf16<GM_BN | GM_RES><<<gr, bl, 0, stream>>>(A, a_bs, lda_m, lda_k,
            Bm, b_bs, ldb_k, ldb_n, M, Nc, K, O, o_bs, ldo, Obf, bias, g, be, res, rel);
        break;
      default:
        gemm_wmma_bf16<GM_DIST><<<gr, bl, 0, stream>>>(A, a_bs, lda_m, lda_k, Bm, b_bs,
            ldb_k, ldb_n, M, Nc, K, O, o_bs, ldo, Obf, bias, g, be, res, rel);
        break;
    }
  };

  // ---- Stem ----
  {
    size_t tot = (size_t)Bn * 24 * 128 * 128;
    conv3x3_bn<<<(tot + 255) / 256, 256, 0, stream>>>(X, st0.w, st0.b, st0.g, st0.be,
        A1, nullptr, Bn, 3, 256, 256, 24, 128, 128, 2, 1);
    tot = (size_t)Bn * 48 * 64 * 64;
    conv3x3_bn<<<(tot + 255) / 256, 256, 0, stream>>>(A1, st1.w, st1.b, st1.g, st1.be,
        A0, nullptr, Bn, 24, 128, 128, 48, 64, 64, 2, 1);
    conv3x3_bn<<<(tot + 255) / 256, 256, 0, stream>>>(A0, st2.w, st2.b, st2.g, st2.be,
        A1, nullptr, Bn, 48, 64, 64, 48, 64, 64, 1, 0);
    add_bcast<<<(tot + 255) / 256, 256, 0, stream>>>(A1, Hb1, pos,
        (size_t)48 * 64 * 64, tot);
  }
  float*  cur  = A1;  float*  alt  = A0;
  __bf16* curb = Hb1; __bf16* altb = Hb0;

  // ---- Modules ----
  struct ModDesc { int type, cin, cout, c, dil, r, hw; };
  const ModDesc mods[15] = {
    {1, 0, 0, 48, 1, 4, 64}, {1, 0, 0, 48, 1, 4, 64},
    {0, 48, 96, 0, 0, 0, 32},
    {1, 0, 0, 96, 1, 2, 32}, {1, 0, 0, 96, 1, 2, 32},
    {0, 96, 240, 0, 0, 0, 16},
    {1, 0, 0, 240, 2, 1, 16}, {1, 0, 0, 240, 2, 1, 16},
    {1, 0, 0, 240, 2, 1, 16}, {1, 0, 0, 240, 2, 1, 16},
    {1, 0, 0, 240, 3, 1, 16}, {1, 0, 0, 240, 3, 1, 16},
    {0, 240, 384, 0, 0, 0, 8},
    {1, 0, 0, 384, 3, 1, 8}, {1, 0, 0, 384, 3, 1, 8},
  };

  float* outp = (float*)d_out;
  const size_t out_off[4] = {0, 1572864, 2359296, 2850816};
  const size_t out_len[4] = {1572864, 786432, 491520, 196608};
  int out_n = 0;

  for (int i = 0; i < 15; ++i) {
    const ModDesc& md = mods[i];
    if (md.type == 0) {
      CP cv = getCP();
      int Hin = md.hw * 2;
      size_t tot = (size_t)Bn * md.cout * md.hw * md.hw;
      conv3x3_bn<<<(tot + 255) / 256, 256, 0, stream>>>(cur, cv.w, cv.b, cv.g, cv.be,
          alt, altb, Bn, md.cin, Hin, Hin, md.cout, md.hw, md.hw, 2, 0);
      float* t = cur; cur = alt; alt = t;
      __bf16* tb = curb; curb = altb; altb = tb;
    } else {
      CP fc1 = getCP(), gcv = getCP(), fc2 = getCP();
      const float* rel = F(p++);
      CP c1 = getCP(), c2 = getCP();

      const int C = md.c, HWd = md.hw, N = HWd * HWd;
      const int rr = md.r, Mp = N / (rr * rr), dil = md.dil;
      const long CN = (long)C * N;

      // fc1: curb(bf16) -> S1 (xf, f32), fused BN
      gemm(GM_BN, fc1.w, 0, C, 1, curb, CN, N, 1, C, N, C, S1, CN, N, nullptr,
           fc1.b, fc1.g, fc1.be, nullptr, nullptr);

      // pooled yf (f32)
      const float* yf = S1;
      if (rr > 1) {
        size_t tot = (size_t)Bn * C * Mp;
        avgpool_rr<<<(tot + 255) / 256, 256, 0, stream>>>(S1, S3, Bn, C, HWd, HWd, rr);
        yf = S3;
      }

      // L2 normalize: xn -> S4 (f32, GEMM A), yn -> S5b (bf16, GEMM B)
      {
        size_t tn = (size_t)Bn * N;
        l2norm_cols<<<(tn + 255) / 256, 256, 0, stream>>>(S1, S4, nullptr, Bn, C, N);
        size_t tm = (size_t)Bn * Mp;
        l2norm_cols<<<(tm + 255) / 256, 256, 0, stream>>>(yf, nullptr, S5b, Bn, C, Mp);
      }

      // dist[n,m] = 2*xn.yn - 2 + rel (GEMM; A = xn^T via lda_m==1 path)
      gemm(GM_DIST, S4, CN, 1, N, S5b, (long)C * Mp, Mp, 1,
           N, Mp, C, Dst, (long)N * Mp, Mp, nullptr,
           nullptr, nullptr, nullptr, nullptr, rel);

      // top-k per row -> 9 neighbor indices
      {
        int rows = Bn * N;
        topk_rows<<<(rows + 7) / 8, 256, 0, stream>>>(Dst, rows, Mp, dil, Idx);
      }

      // gather + max aggregation -> interleaved bf16 (B,2C,N) in S2b
      {
        dim3 gg(N / 32, 1, Bn);
        gather_agg<<<gg, 256, 0, stream>>>(S1, yf, Idx, S2b, C, N, Mp);
      }

      // gconv (2C->2C) BN+GELU: S2b -> alt(f32) + Gb(bf16)
      gemm(GM_BN | GM_GELU, gcv.w, 0, 2 * C, 1, S2b, (long)2 * C * N, N, 1,
           2 * C, N, 2 * C, alt, (long)2 * C * N, N, Gb,
           gcv.b, gcv.g, gcv.be, nullptr, nullptr);

      // fc2 (2C->C) BN + residual(cur): Gb -> S1(f32) + S1b(bf16)  (h')
      gemm(GM_BN | GM_RES, fc2.w, 0, 2 * C, 1, Gb, (long)2 * C * N, N, 1,
           C, N, 2 * C, S1, CN, N, S1b, fc2.b, fc2.g, fc2.be, cur, nullptr);

      // FFN c1 (C->4C) BN+GELU: S1b -> S2(f32) + S2b(bf16)
      gemm(GM_BN | GM_GELU, c1.w, 0, C, 1, S1b, CN, N, 1,
           4 * C, N, C, S2, (long)4 * C * N, N, S2b,
           c1.b, c1.g, c1.be, nullptr, nullptr);

      // FFN c2 (4C->C) BN + residual(S1): S2b -> alt(f32) + altb(bf16)
      gemm(GM_BN | GM_RES, c2.w, 0, 4 * C, 1, S2b, (long)4 * C * N, N, 1,
           C, N, 4 * C, alt, CN, N, altb, c2.b, c2.g, c2.be, S1, nullptr);

      float* t = cur; cur = alt; alt = t;
      __bf16* tb = curb; curb = altb; altb = tb;
    }

    if (i == 1 || i == 4 || i == 11 || i == 14) {
      (void)hipMemcpyAsync(outp + out_off[out_n], cur,
                           out_len[out_n] * sizeof(float),
                           hipMemcpyDeviceToDevice, stream);
      ++out_n;
    }
  }
}
